// OlmoAttention_69861938036889
// MI455X (gfx1250) — matrix operand order repaired and checked by reading, not compiled, not run on hardware
//
#include <hip/hip_runtime.h>

typedef __attribute__((ext_vector_type(16))) _Float16 v16h;
typedef __attribute__((ext_vector_type(8)))  _Float16 v8h;
typedef __attribute__((ext_vector_type(8)))  float    v8f;
typedef __attribute__((ext_vector_type(4)))  unsigned u32x4;
typedef __attribute__((ext_vector_type(4)))  int      i32x4;
typedef __attribute__((ext_vector_type(8)))  int      i32x8;

#define B_   2
#define S_   2048
#define H_   4096
#define NH_  32
#define HD_  128
#define N3_  (3 * H_)
#define MROWS_ (B_ * S_)
#define SCALE_ 0.08838834764831845f   // 128^-0.5
#define LOG_THETA_ 9.210340371976184f // ln(10000)

// ---------------------------------------------------------------------------
// Fragment loaders (wave32); A and B share this code's contiguous K order
// A (16x32 f16): lane L<16 -> row L, K = kb+0..15 contiguous
//                lane L>=16 -> row L-16, K = kb+16..31 contiguous
// B (32x16 f16): lane n<16 -> col n, K = kb+0..15 contiguous
//                lane n>=16 -> col n-16, K = kb+16..31 contiguous
// C/D (16x16 f32): VGPR v: lanes 0-15 row v, lanes 16-31 row v+8; col = lane&15
// ---------------------------------------------------------------------------
__device__ inline v16h load_a_frag(const _Float16* __restrict__ A, int ld,
                                   int row0, int kb) {
  const int lane = threadIdx.x & 31;
  const int l = lane & 15;
  const int hi = lane >> 4;
  const _Float16* p = A + (long)(row0 + l) * ld + kb + hi * 16;
  v8h c0 = *(const v8h*)(p);
  v8h c1 = *(const v8h*)(p + 8);
  v16h r;
#pragma unroll
  for (int i = 0; i < 8; ++i) { r[i] = c0[i]; r[8 + i] = c1[i]; }
  return r;
}

__device__ inline v16h load_b_frag(const _Float16* __restrict__ W, int ld,
                                   int n0, int kb) {
  const int lane = threadIdx.x & 31;
  const _Float16* p = W + (long)(n0 + (lane & 15)) * ld + kb + (lane >> 4) * 16;
  return *(const v16h*)(p);
}

// ---------------------------------------------------------------------------
// TDM: async DMA of a 256(N) x 32(K) f16 weight tile into LDS.
// 2D descriptor per CDNA5 ISA:
//   group0: count=1 | lds_addr | global_addr(57b) | type=2
//   group1: data_size=1(2B), tensor_dim0=K, tensor_dim1=N,
//           tile_dim0=32, tile_dim1=256, tile_dim2=0(unused),
//           tensor_dim0_stride=K, dim1_stride unused for 2D
// ---------------------------------------------------------------------------
__device__ inline void tdm_load_b_tile(const _Float16* W, unsigned N, unsigned K,
                                       long nblk, int kb, unsigned lds_off) {
  unsigned long long ga =
      (unsigned long long)(const void*)(W + nblk * (long)K + kb);
  u32x4 g0 = { 1u,                                   // count=1, user mode
               lds_off,                              // LDS byte address
               (unsigned)ga,                         // global_addr[31:0]
               (unsigned)((ga >> 32) & 0x01FFFFFFu) | 0x80000000u }; // type=2
  i32x8 g1 = { (int)(1u << 16),                      // data_size = 2 bytes
               (int)((K & 0xFFFFu) << 16),           // tensor_dim0 lo16
               (int)((K >> 16) | ((N & 0xFFFFu) << 16)), // dim0 hi16 | dim1 lo16
               (int)((N >> 16) | (32u << 16)),       // dim1 hi16 | tile_dim0=32
               (int)256u,                            // tile_dim1=256, tile_dim2=0
               (int)K,                               // tensor_dim0_stride lo32
               0, 0 };                               // stride hi bits, dim1_stride
  i32x4 z4 = { 0, 0, 0, 0 };
#if defined(__clang_major__) && (__clang_major__ >= 23)
  i32x8 z8 = { 0, 0, 0, 0, 0, 0, 0, 0 };
  __builtin_amdgcn_tensor_load_to_lds(g0, g1, z4, z4, z8, 0);
#else
  __builtin_amdgcn_tensor_load_to_lds(g0, g1, z4, z4, 0);
#endif
}

// ---------------------------------------------------------------------------
// fp32 -> f16 conversion (4 elements / thread)
// ---------------------------------------------------------------------------
__global__ __launch_bounds__(256) void f32_to_f16(const float* __restrict__ x,
                                                  _Float16* __restrict__ y,
                                                  long n) {
  long i = ((long)blockIdx.x * blockDim.x + threadIdx.x) * 4;
  if (i + 3 < n) {
#pragma unroll
    for (int j = 0; j < 4; ++j) y[i + j] = (_Float16)x[i + j];
  }
}

// ---------------------------------------------------------------------------
// GEMM: out[M,N] = A[M,K] (f16, row-major) @ W[N,K]^T (f16, row-major)
// Block = 8 waves -> 32(M) x 256(N) tile; each wave 16x64.
// Weight tiles are TDM-DMA'd into double-buffered LDS; A streams via L2.
// CLIP: clamp to +-8 and emit f16; else emit f32.
// ---------------------------------------------------------------------------
template <bool CLIP_F16_OUT>
__global__ __launch_bounds__(256) void gemm_wmma(const _Float16* __restrict__ A,
                                                 const _Float16* __restrict__ W,
                                                 void* __restrict__ out,
                                                 int M, int N, int K) {
  __shared__ __align__(64) _Float16 shB[2][256 * 32];
  const int lane = threadIdx.x & 31;
  const int wave = threadIdx.x >> 5;
  const int m0 = (blockIdx.y * 2 + (wave & 1)) * 16;
  const long nblk = (long)blockIdx.x * 256;
  const int nw = (wave >> 1) * 64;  // this wave's column offset inside the tile

  typedef __attribute__((address_space(3))) _Float16 lds_f16;
  unsigned lds_off[2];
  lds_off[0] = (unsigned)(unsigned long long)(lds_f16*)(&shB[0][0]);
  lds_off[1] = (unsigned)(unsigned long long)(lds_f16*)(&shB[1][0]);

  const int niter = K >> 5;
  if (wave == 0) tdm_load_b_tile(W, (unsigned)N, (unsigned)K, nblk, 0, lds_off[0]);

  v8f acc[4] = {};
  for (int i = 0; i < niter; ++i) {
    const int kb = i << 5;
    const int buf = i & 1;
    if (wave == 0) __builtin_amdgcn_s_wait_tensorcnt(0);
    __syncthreads();  // buf ready for all waves; buf^1 fully consumed
    if (wave == 0 && (i + 1) < niter)
      tdm_load_b_tile(W, (unsigned)N, (unsigned)K, nblk, kb + 32, lds_off[buf ^ 1]);

    v16h a = load_a_frag(A, K, m0, kb);
#pragma unroll
    for (int t = 0; t < 4; ++t) {
      // B fragment from LDS tile: lane = local column, 16 contiguous K halves
      const _Float16* bp =
          &shB[buf][(nw + t * 16 + (lane & 15)) * 32 + (lane >> 4) * 16];
      v16h b = *(const v16h*)bp;
      acc[t] = __builtin_amdgcn_wmma_f32_16x16x32_f16(false, a, false, b,
                                                      (short)0, acc[t], false, false);
    }
  }

  const int col0 = lane & 15;
  const int rofs = (lane < 16) ? 0 : 8;
#pragma unroll
  for (int t = 0; t < 4; ++t) {
#pragma unroll
    for (int v = 0; v < 8; ++v) {
      long row = m0 + rofs + v;
      long col = nblk + nw + t * 16 + col0;
      float x = acc[t][v];
      if constexpr (CLIP_F16_OUT) {
        x = fminf(fmaxf(x, -8.f), 8.f);
        ((_Float16*)out)[row * N + col] = (_Float16)x;
      } else {
        ((float*)out)[row * N + col] = x;
      }
    }
  }
}

// ---------------------------------------------------------------------------
// RoPE + split + V transpose.
// qkv f16 [B*S, 3H] -> Q,K f16 [B,NH,S,HD] (roped), Vt f16 [B,NH,HD,S]
// One thread per (b, s, nh, i) with i in [0,64): handles pair (i, i+64).
// ---------------------------------------------------------------------------
__global__ __launch_bounds__(256) void rope_split(const _Float16* __restrict__ qkv,
                                                  const int* __restrict__ pos,
                                                  _Float16* __restrict__ Q,
                                                  _Float16* __restrict__ K,
                                                  _Float16* __restrict__ Vt) {
  long idx = (long)blockIdx.x * blockDim.x + threadIdx.x;
  int i  = (int)(idx & 63);
  int nh = (int)((idx >> 6) & (NH_ - 1));
  int s  = (int)((idx >> 11) & (S_ - 1));
  int b  = (int)(idx >> 22);
  if (b >= B_) return;

  float p   = (float)pos[s];
  float inv = __expf(-(float)i * (LOG_THETA_ / 64.f));
  float ang = p * inv;
  float c = __cosf(ang), sn = __sinf(ang);

  long row = (long)(b * S_ + s) * N3_;
  int  ch  = nh * HD_ + i;

  float q1 = (float)qkv[row + ch];
  float q2 = (float)qkv[row + ch + 64];
  float k1 = (float)qkv[row + H_ + ch];
  float k2 = (float)qkv[row + H_ + ch + 64];
  float v1 = (float)qkv[row + 2 * H_ + ch];
  float v2 = (float)qkv[row + 2 * H_ + ch + 64];

  long qbase = ((long)(b * NH_ + nh) * S_ + s) * HD_;
  Q[qbase + i]      = (_Float16)(q1 * c - q2 * sn);
  Q[qbase + i + 64] = (_Float16)(q2 * c + q1 * sn);
  K[qbase + i]      = (_Float16)(k1 * c - k2 * sn);
  K[qbase + i + 64] = (_Float16)(k2 * c + k1 * sn);

  long vbase = (long)(b * NH_ + nh) * HD_ * S_;
  Vt[vbase + (long)i * S_ + s]        = (_Float16)v1;
  Vt[vbase + (long)(i + 64) * S_ + s] = (_Float16)v2;
}

// ---------------------------------------------------------------------------
// Causal flash attention. One wave per 16-query tile; streams 32-key blocks
// with online softmax. P goes f32(C-layout) -> LDS -> f16 A-fragment.
// Output: attn f16 [B*S, H] (h = nh*HD + hd) for the final projection.
// ---------------------------------------------------------------------------
__global__ __launch_bounds__(256) void flash_attn(const _Float16* __restrict__ Q,
                                                  const _Float16* __restrict__ K,
                                                  const _Float16* __restrict__ Vt,
                                                  _Float16* __restrict__ Aout) {
  __shared__ __align__(64) _Float16 lds[8][16 * 32];
  const int lane = threadIdx.x & 31;
  const int wave = threadIdx.x >> 5;

  long wid = (long)blockIdx.x * 8 + wave;
  int qt = (int)(wid & (S_ / 16 - 1));      // 128 q-tiles per head
  long hb = wid >> 7;
  int h = (int)(hb & (NH_ - 1));
  int b = (int)(hb >> 5);
  int q0 = qt * 16;

  const _Float16* Qh = Q + (long)(b * NH_ + h) * S_ * HD_;
  const _Float16* Kh = K + (long)(b * NH_ + h) * S_ * HD_;
  const _Float16* Vh = Vt + (long)(b * NH_ + h) * HD_ * S_;

  v16h qf[4];
#pragma unroll
  for (int t = 0; t < 4; ++t) qf[t] = load_a_frag(Qh, HD_, q0, t * 32);

  v8f o[8] = {};
  float m[8], l[8];
#pragma unroll
  for (int v = 0; v < 8; ++v) { m[v] = -1e30f; l[v] = 0.f; }

  const int col0 = lane & 15;
  const int rofs = (lane < 16) ? 0 : 8;

  for (int kbase = 0; kbase < q0 + 16; kbase += 32) {
    float p[2][8];
    float mt[8];
#pragma unroll
    for (int v = 0; v < 8; ++v) mt[v] = -1e30f;

    // scores: Q @ K^T for two 16-key tiles
#pragma unroll
    for (int t = 0; t < 2; ++t) {
      v8f s = {};
#pragma unroll
      for (int kk = 0; kk < 4; ++kk) {
        v16h bf = load_b_frag(Kh, HD_, kbase + t * 16, kk * 32);
        s = __builtin_amdgcn_wmma_f32_16x16x32_f16(false, qf[kk], false, bf,
                                                   (short)0, s, false, false);
      }
      int col = kbase + t * 16 + col0;
#pragma unroll
      for (int v = 0; v < 8; ++v) {
        int row = q0 + rofs + v;
        float x = (col <= row) ? s[v] * SCALE_ : -1e30f;
        p[t][v] = x;
        mt[v] = fmaxf(mt[v], x);
      }
    }

    // row-max across the 16-lane half-wave
#pragma unroll
    for (int off = 1; off < 16; off <<= 1)
#pragma unroll
      for (int v = 0; v < 8; ++v)
        mt[v] = fmaxf(mt[v], __shfl_xor(mt[v], off, 32));

    float alpha[8];
#pragma unroll
    for (int v = 0; v < 8; ++v) {
      float mn = fmaxf(m[v], mt[v]);
      alpha[v] = __expf(m[v] - mn);
      m[v] = mn;
    }

    float rs[8];
#pragma unroll
    for (int v = 0; v < 8; ++v) {
      p[0][v] = __expf(p[0][v] - m[v]);
      p[1][v] = __expf(p[1][v] - m[v]);
      rs[v] = p[0][v] + p[1][v];
    }
#pragma unroll
    for (int off = 1; off < 16; off <<= 1)
#pragma unroll
      for (int v = 0; v < 8; ++v)
        rs[v] += __shfl_xor(rs[v], off, 32);
#pragma unroll
    for (int v = 0; v < 8; ++v) l[v] = l[v] * alpha[v] + rs[v];

#pragma unroll
    for (int nt = 0; nt < 8; ++nt)
#pragma unroll
      for (int v = 0; v < 8; ++v) o[nt][v] *= alpha[v];

    // C-layout probs -> LDS (row-major 16x32 f16)
#pragma unroll
    for (int t = 0; t < 2; ++t)
#pragma unroll
      for (int v = 0; v < 8; ++v)
        lds[wave][(rofs + v) * 32 + t * 16 + col0] = (_Float16)p[t][v];
    asm volatile("s_wait_dscnt 0" ::: "memory");

    // reload as A-fragment (16x32)
    v16h pf;
    {
      int lr = lane & 15, hi = (lane >> 4) * 16;
#pragma unroll
      for (int i = 0; i < 8; ++i) {
        pf[i]     = lds[wave][lr * 32 + hi + i];
        pf[8 + i] = lds[wave][lr * 32 + hi + 8 + i];
      }
    }

    // P @ V : Vt gives contiguous-along-keys B fragments
#pragma unroll
    for (int nt = 0; nt < 8; ++nt) {
      const _Float16* vp = Vh + (long)(nt * 16 + col0) * S_ + kbase + (lane >> 4) * 16;
      v16h bf = *(const v16h*)vp;
      o[nt] = __builtin_amdgcn_wmma_f32_16x16x32_f16(false, pf, false, bf,
                                                     (short)0, o[nt], false, false);
    }
  }

  // normalize and store f16 [B*S, H]
#pragma unroll
  for (int v = 0; v < 8; ++v) {
    float inv = 1.f / l[v];
    long row = (long)(b * S_ + q0 + rofs + v) * H_ + h * HD_;
#pragma unroll
    for (int nt = 0; nt < 8; ++nt)
      Aout[row + nt * 16 + col0] = (_Float16)(o[nt][v] * inv);
  }
}

// ---------------------------------------------------------------------------
extern "C" void kernel_launch(void* const* d_in, const int* in_sizes, int n_in,
                              void* d_out, int out_size, void* d_ws, size_t ws_size,
                              hipStream_t stream) {
  const float* x    = (const float*)d_in[0];
  const int*   pos  = (const int*)d_in[1];
  const float* wqkv = (const float*)d_in[2];
  const float* wo   = (const float*)d_in[3];

  char* ws = (char*)d_ws;
  _Float16* Xh    = (_Float16*)ws; ws += (size_t)MROWS_ * H_ * 2;   // 33.5 MB
  _Float16* Wqkvh = (_Float16*)ws; ws += (size_t)N3_ * H_ * 2;      // 100.7 MB
  _Float16* Woh   = (_Float16*)ws; ws += (size_t)H_ * H_ * 2;       // 33.5 MB
  _Float16* qkvh  = (_Float16*)ws; ws += (size_t)MROWS_ * N3_ * 2;  // 100.7 MB
  _Float16* Qf    = (_Float16*)ws; ws += (size_t)MROWS_ * H_ * 2;   // 33.5 MB
  _Float16* Kf    = (_Float16*)ws; ws += (size_t)MROWS_ * H_ * 2;   // 33.5 MB
  _Float16* Vtf   = (_Float16*)ws; ws += (size_t)MROWS_ * H_ * 2;   // 33.5 MB
  _Float16* Atn   = (_Float16*)ws; ws += (size_t)MROWS_ * H_ * 2;   // 33.5 MB

  // 1) convert fp32 -> f16
  {
    long n1 = (long)MROWS_ * H_;
    long n2 = (long)N3_ * H_;
    f32_to_f16<<<(int)(n1 / 1024), 256, 0, stream>>>(x, Xh, n1);
    f32_to_f16<<<(int)(n2 / 1024), 256, 0, stream>>>(wqkv, Wqkvh, n2);
    f32_to_f16<<<(int)(n1 / 1024), 256, 0, stream>>>(wo, Woh, n1);
  }

  // 2) QKV projection (clip, f16 out): [4096,12288] = Xh @ Wqkv^T
  gemm_wmma<true><<<dim3(N3_ / 256, MROWS_ / 32), 256, 0, stream>>>(
      Xh, Wqkvh, (void*)qkvh, MROWS_, N3_, H_);

  // 3) RoPE + head split + V transpose
  rope_split<<<(int)((long)B_ * S_ * NH_ * 64 / 256), 256, 0, stream>>>(
      qkvh, pos, Qf, Kf, Vtf);

  // 4) causal flash attention: 8192 waves, 8 waves/block
  flash_attn<<<B_ * NH_ * (S_ / 16) / 8, 256, 0, stream>>>(Qf, Kf, Vtf, Atn);

  // 5) output projection (f32 out): d_out = Atn @ Wo^T
  gemm_wmma<false><<<dim3(H_ / 256, MROWS_ / 32), 256, 0, stream>>>(
      Atn, Woh, d_out, MROWS_, H_, H_);
}
